// metricLoss_80659485818924
// MI455X (gfx1250) — matrix operand order repaired and checked by reading, not compiled, not run on hardware
//
#include <hip/hip_runtime.h>

typedef __attribute__((ext_vector_type(16))) _Float16 v16h;
typedef __attribute__((ext_vector_type(8)))  _Float16 v8h;
typedef __attribute__((ext_vector_type(4)))  _Float16 v4h;
typedef __attribute__((ext_vector_type(8)))  float    v8f;
typedef __attribute__((ext_vector_type(4)))  float    v4f;

#define Dm   64                 // embedding dim
#define Kc   32                 // clusters
#define Np   (512*512)          // pixels per image
#define TP   128                // pixels per LDS tile
#define WGPI 128                // workgroups per image
#define TPW  (Np/(TP*WGPI))     // tiles per workgroup = 16
#define PART 2112               // per-wg partial: 64*32 S + 32 cnt + 32 sq
#define FSTR 136                // ldsF row stride (halves), 272B = 16B-aligned
#define OSTR 136                // ldsOH row stride (halves)

__device__ inline v16h cat16(v8h a, v8h b) {
    v16h r;
#pragma unroll
    for (int i = 0; i < 8; ++i) { r[i] = a[i]; r[i + 8] = b[i]; }
    return r;
}

// ---------------------------------------------------------------------------
// Kernel 1: streaming segmented reduction via one-hot WMMA GEMM.
// Each workgroup (128 threads = 4 waves) owns TPW tiles of TP pixels and
// produces a partial (S[64][32], cnt[32], Sq[32]).
// ---------------------------------------------------------------------------
__global__ __launch_bounds__(128) void seg_reduce_kernel(
    const float* __restrict__ feats,   // (B, 64, N) pixel-major
    const int*   __restrict__ labels,  // (B, N)
    float*       __restrict__ partials)
{
    __shared__ _Float16 ldsF[64 * FSTR];    // f16 feature tile (dims x pixels)
    __shared__ _Float16 ldsOH[32 * OSTR];   // one-hot tile (clusters x pixels)
    __shared__ int      ldsLab[TP];
    __shared__ float    ldsSqP[4 * TP];     // per-dim-block squared partials
    __shared__ float    cntAcc[Kc];
    __shared__ float    sqAcc[Kc];

    const int t    = threadIdx.x;
    const int lane = t & 31;
    const int w    = t >> 5;          // wave id 0..3
    const int wg   = blockIdx.x;
    const int b    = wg / WGPI;
    const int wgi  = wg % WGPI;

    const float* Fb = feats  + (size_t)b * Dm * Np;
    const int*   Lb = labels + (size_t)b * Np;

    if (t < Kc) { cntAcc[t] = 0.f; sqAcc[t] = 0.f; }

    v8f acc[4][2];
#pragma unroll
    for (int mt = 0; mt < 4; ++mt)
#pragma unroll
        for (int nt = 0; nt < 2; ++nt) acc[mt][nt] = v8f{};

    const int g   = t & 31;    // pixel group (4 px) within tile
    const int blk = t >> 5;    // dim block (16 dims)

    for (int tile = 0; tile < TPW; ++tile) {
        const int tb = (wgi * TPW + tile) * TP;
        __syncthreads();   // previous tile's WMMA readers done; also orders init

        // --- stage A: clear one-hot slice, load labels, stage f16 feature tile
        {
            _Float16* c0 = &ldsOH[(t >> 2) * OSTR + (t & 3) * 32];
            v8h z{};
            *(v8h*)(c0 + 0) = z; *(v8h*)(c0 + 8)  = z;
            *(v8h*)(c0 + 16) = z; *(v8h*)(c0 + 24) = z;
        }
        ldsLab[t] = Lb[tb + t];

        float sq0 = 0.f, sq1 = 0.f, sq2 = 0.f, sq3 = 0.f;
#pragma unroll
        for (int i = 0; i < 16; ++i) {
            const int d = blk * 16 + i;
            const float* src = Fb + (size_t)d * Np + tb + g * 4;
            const v4f v = *(const v4f*)src;
            if (tile + 1 < TPW) __builtin_prefetch(src + TP, 0, 1);
            sq0 += v.x * v.x; sq1 += v.y * v.y;
            sq2 += v.z * v.z; sq3 += v.w * v.w;
            v4h h; h[0] = (_Float16)v.x; h[1] = (_Float16)v.y;
                   h[2] = (_Float16)v.z; h[3] = (_Float16)v.w;
            *(v4h*)&ldsF[d * FSTR + g * 4] = h;
        }
        ldsSqP[blk * TP + g * 4 + 0] = sq0;
        ldsSqP[blk * TP + g * 4 + 1] = sq1;
        ldsSqP[blk * TP + g * 4 + 2] = sq2;
        ldsSqP[blk * TP + g * 4 + 3] = sq3;
        __syncthreads();

        // --- stage B: per-pixel owner sets one-hot, accumulates cnt / Sq
        {
            const int   lab = ldsLab[t];
            const float sq  = ldsSqP[0 * TP + t] + ldsSqP[1 * TP + t] +
                              ldsSqP[2 * TP + t] + ldsSqP[3 * TP + t];
            atomicAdd(&cntAcc[lab], 1.0f);
            atomicAdd(&sqAcc[lab], sq);
            ldsOH[lab * OSTR + t] = (_Float16)1.0f;
        }
        __syncthreads();

        // --- compute: wave w handles the 32-pixel chunk p..p+31 (K dim of WMMA)
        const int p  = 32 * w;
        const int hi = lane >> 4;       // wave half
        const int r  = lane & 15;

        v16h bf[2];
#pragma unroll
        for (int nt = 0; nt < 2; ++nt) {
            // B (32x16 one-hot): lanes 0-15 hold K=0..15, lanes 16-31 K=16..31
            const _Float16* src = &ldsOH[(r + 16 * nt) * OSTR + p + hi * 16];
            bf[nt] = cat16(*(const v8h*)src, *(const v8h*)(src + 8));
        }
#pragma unroll
        for (int mt = 0; mt < 4; ++mt) {
            // A (16x32 f16): lanes 0-15 hold K=0..15, lanes 16-31 K=16..31
            const _Float16* srcA = &ldsF[(mt * 16 + r) * FSTR + p + hi * 16];
            const v16h af = cat16(*(const v8h*)srcA, *(const v8h*)(srcA + 8));
            acc[mt][0] = __builtin_amdgcn_wmma_f32_16x16x32_f16(
                false, af, false, bf[0], (short)0, acc[mt][0], false, false);
            acc[mt][1] = __builtin_amdgcn_wmma_f32_16x16x32_f16(
                false, af, false, bf[1], (short)0, acc[mt][1], false, false);
        }
    }

    // --- cross-wave reduction of S into LDS (reuse ldsF as f32 scratch)
    float* ldsS = (float*)ldsF;        // 64*33 floats = 8448B <= ldsF size
    __syncthreads();
    for (int e = t; e < 64 * 33; e += 128) ldsS[e] = 0.f;
    __syncthreads();
    {
        const int kbase = lane & 15;
        const int dofs  = (lane >> 4) << 3;   // C-layout: VGPR j -> M=j or j+8
#pragma unroll
        for (int mt = 0; mt < 4; ++mt)
#pragma unroll
            for (int nt = 0; nt < 2; ++nt)
#pragma unroll
                for (int j = 0; j < 8; ++j) {
                    const int d = mt * 16 + j + dofs;
                    const int k = kbase + 16 * nt;
                    atomicAdd(&ldsS[d * 33 + k], acc[mt][nt][j]);
                }
    }
    __syncthreads();

    float* outp = partials + (size_t)wg * PART;
    for (int e = t; e < PART; e += 128) {
        float v;
        if (e < 2048)      v = ldsS[(e >> 5) * 33 + (e & 31)];
        else if (e < 2080) v = cntAcc[e - 2048];
        else               v = sqAcc[e - 2080];
        outp[e] = v;
    }
}

// ---------------------------------------------------------------------------
// Kernel 2: per-image reduction of partials + K x K loss math.
// ---------------------------------------------------------------------------
__global__ __launch_bounds__(128) void loss_kernel(
    const float* __restrict__ partials, float* __restrict__ lossb)
{
    __shared__ float Sl[PART];
    __shared__ float mu[Kc][68];
    __shared__ float cntS[Kc], safeS[Kc], m2S[Kc], varS[Kc], presS[Kc];
    __shared__ float red[128];

    const int t = threadIdx.x;
    const int b = blockIdx.x;
    const float* base = partials + (size_t)b * WGPI * PART;

    for (int e = t; e < PART; e += 128) {
        float s = 0.f;
        for (int p = 0; p < WGPI; ++p) s += base[(size_t)p * PART + e];
        Sl[e] = s;
    }
    __syncthreads();

    if (t < Kc) {
        const float c = Sl[2048 + t];
        cntS[t]  = c;
        presS[t] = (c > 0.f) ? 1.f : 0.f;
        safeS[t] = fmaxf(c, 1.f);
    }
    __syncthreads();

    for (int e = t; e < 2048; e += 128) {
        const int d = e >> 5, k = e & 31;
        mu[k][d] = Sl[e] / safeS[k];
    }
    __syncthreads();

    if (t < Kc) {
        float m2 = 0.f;
        for (int d = 0; d < Dm; ++d) { const float x = mu[t][d]; m2 += x * x; }
        m2S[t] = m2;
        // sum_{i in k} ||f_i - mu_k||^2 = Sq_k - cnt_k * ||mu_k||^2
        const float var = (Sl[2080 + t] - cntS[t] * m2) / safeS[t];
        varS[t] = (presS[t] > 0.f) ? var : 0.f;
    }
    __syncthreads();

    float h = 0.f;
    for (int idx = t; idx < Kc * Kc; idx += 128) {
        const int i = idx >> 5, j = idx & 31;
        if (i < j && presS[i] > 0.f && presS[j] > 0.f) {
            float dsq = 0.f;
            for (int d = 0; d < Dm; ++d) {
                const float df = mu[i][d] - mu[j][d];
                dsq += df * df;
            }
            const float dist = sqrtf(dsq);
            if (dist < 3.0f) { const float m = 3.0f - dist; h += m * m; }
        }
    }
    red[t] = h;
    __syncthreads();

    if (t == 0) {
        float hinge = 0.f;
        for (int i = 0; i < 128; ++i) hinge += red[i];
        float var = 0.f, reg = 0.f, ncl = 0.f;
        for (int k = 0; k < Kc; ++k) {
            var += varS[k];
            ncl += presS[k];
            if (presS[k] > 0.f) reg += sqrtf(m2S[k]);
        }
        const float dist_loss = hinge / fmaxf(ncl - 1.f, 1.f);
        lossb[b] = (var + dist_loss + 0.001f * reg) / ncl;
    }
}

__global__ void final_kernel(const float* __restrict__ lossb, float* __restrict__ out, int B)
{
    if (threadIdx.x == 0) {
        float s = 0.f;
        for (int b = 0; b < B; ++b) s += lossb[b];
        out[0] = s / (float)(B + 1);
    }
}

// ---------------------------------------------------------------------------
extern "C" void kernel_launch(void* const* d_in, const int* in_sizes, int n_in,
                              void* d_out, int out_size, void* d_ws, size_t ws_size,
                              hipStream_t stream)
{
    const float* feats  = (const float*)d_in[0];
    const int*   labels = (const int*)d_in[1];
    float*       out    = (float*)d_out;

    const int B = in_sizes[0] / (Dm * Np);   // = 4 for the reference shapes

    float* partials = (float*)d_ws;                       // B*WGPI*PART floats
    float* lossb    = partials + (size_t)B * WGPI * PART; // B floats

    seg_reduce_kernel<<<B * WGPI, 128, 0, stream>>>(feats, labels, partials);
    loss_kernel<<<B, 128, 0, stream>>>(partials, lossb);
    final_kernel<<<1, 32, 0, stream>>>(lossb, out, B);
}
